// time_combination_nn2_75720273428862
// MI455X (gfx1250) — matrix operand order repaired and checked by reading, not compiled, not run on hardware
//
#include <hip/hip_runtime.h>
#include <hip/hip_bf16.h>
#include <math.h>

typedef __attribute__((ext_vector_type(16))) _Float16 v16h;
typedef __attribute__((ext_vector_type(8)))  float    v8f;

#define BSZ   8192
#define SSZ   1024
#define CSZ   13
#define KPAD  (SSZ * 16)      // 16384: per-s channel dim padded 13 -> 16
#define NPAD  112             // 100 -> 112 (7 WMMA N-tiles)
#define SCHUNK 32             // s-values per outer iteration
#define KCHUNK (SCHUNK * 16)  // 512: K' per outer iteration
#define ASTR  (KCHUNK + 16)   // LDS A-tile row stride in halves

// ---------- prep 1: collapse 13->20->13 (no act in middle) into M (13x13), bb (13) ----------
__global__ void prep_small(const float* __restrict__ W01, const float* __restrict__ b01,
                           const float* __restrict__ W02, const float* __restrict__ b02,
                           float* __restrict__ Mg, float* __restrict__ bbg) {
  int t = threadIdx.x;
  if (t < 169) {
    int c2 = t / 13, cp = t % 13;
    float acc = 0.f;
    for (int j = 0; j < 20; ++j) acc += W02[c2 * 20 + j] * W01[j * 13 + cp];
    Mg[t] = acc;
  }
  if (t < 13) {
    float acc = b02[t];
    for (int j = 0; j < 20; ++j) acc += W02[t * 20 + j] * b01[j];
    bbg[t] = acc;
  }
}

// ---------- prep 2: repack W1a (100 x 13312 f32) -> f16 (112 x 16384), K' = s*16 + c ----------
__global__ void prep_w1a(const float* __restrict__ W1a, _Float16* __restrict__ W1ah) {
  int idx = blockIdx.x * 256 + threadIdx.x;
  if (idx >= NPAD * KPAD) return;
  int r = idx / KPAD;
  int k = idx - r * KPAD;
  int s = k >> 4, c = k & 15;
  float v = 0.f;
  if (r < 100 && c < 13) v = W1a[r * (SSZ * CSZ) + s * 13 + c];
  W1ah[idx] = (_Float16)v;
}

// ---------- main fused kernel: 1 block = 16 batch rows ----------
__global__ void __launch_bounds__(256, 1) fused_main(
    const float* __restrict__ x,
    const float* __restrict__ w00, const float* __restrict__ b00,
    const _Float16* __restrict__ W1ah,
    const float* __restrict__ Mg, const float* __restrict__ bbg,
    const float* __restrict__ b1a, const float* __restrict__ W1b,
    const float* __restrict__ b1b,
    float* __restrict__ out)
{
  __shared__ _Float16 Atile[2][16][ASTR]; // double-buffered: 16 batch rows x 512 K' halves
  __shared__ float Mlds[13][16];
  __shared__ float bbld[16];
  __shared__ float ep[7][16][16];         // epilogue partials [ntile][batch][n]

  const int tid    = threadIdx.x;
  const int bl     = tid >> 4;            // 0..15 local batch row
  const int sq     = tid & 15;            // 0..15 s-pair slot
  const int wave   = tid >> 5;            // 0..7
  const int lane   = tid & 31;
  const int batch0 = blockIdx.x * 16;

  if (tid < 169) Mlds[tid / 13][tid % 13] = Mg[tid];
  if (tid < 13)  bbld[tid] = bbg[tid];

  const float* xrow = x + (size_t)(batch0 + bl) * (CSZ * 2 * SSZ);

  // B fragment base: column n = lane&15 of this wave's N-tile is W1a row r (contiguous K).
  const _Float16* Bbase =
      W1ah + (size_t)(((wave < 7 ? wave : 0) * 16) + (lane & 15)) * KPAD + ((lane >> 4) << 4);
  // A fragment base within a row: lane<16 -> K {0..15}; lane>=16 -> +16 halves.
  const int aoff = ((lane >> 4) << 4);
  const int arow = lane & 15;

  v8f acc = {};

  __syncthreads();

  for (int it = 0; it < SSZ / SCHUNK; ++it) {
    const int buf = it & 1;

    // ---- stage 1 + fused 13x13 ReLU stage: each thread computes 2 s-values ----
    const int sA = it * SCHUNK + 2 * sq;
    float2 wA = *(const float2*)(w00 + 2 * sA);
    float2 wB = *(const float2*)(w00 + 2 * sA + 2);
    float baA = b00[sA], baB = b00[sA + 1];

    float oA[13], oB[13];
#pragma unroll
    for (int c = 0; c < 13; ++c) {
      float4 xv = *(const float4*)(xrow + c * (2 * SSZ) + 2 * sA);
      oA[c] = fmaf(xv.x, wA.x, fmaf(xv.y, wA.y, baA));
      oB[c] = fmaf(xv.z, wB.x, fmaf(xv.w, wB.y, baB));
    }
    float hA[13], hB[13];
#pragma unroll
    for (int c = 0; c < 13; ++c) { hA[c] = bbld[c]; hB[c] = bbld[c]; }
#pragma unroll
    for (int cp = 0; cp < 13; ++cp) {
#pragma unroll
      for (int c2 = 0; c2 < 13; ++c2) {
        float m = Mlds[c2][cp];              // uniform LDS broadcast
        hA[c2] = fmaf(m, oA[cp], hA[c2]);
        hB[c2] = fmaf(m, oB[cp], hB[c2]);
      }
    }
    // ReLU + f16 pack: two 16-half K'-slots (channel padded 13->16 with zeros)
    uint4 pk[4];
    _Float16* ph = (_Float16*)pk;
#pragma unroll
    for (int c = 0; c < 16; ++c) {
      ph[c]      = (c < 13) ? (_Float16)fmaxf(hA[c], 0.f) : (_Float16)0.f;
      ph[16 + c] = (c < 13) ? (_Float16)fmaxf(hB[c], 0.f) : (_Float16)0.f;
    }
    uint4* dst = (uint4*)&Atile[buf][bl][sq * 32];
    dst[0] = pk[0]; dst[1] = pk[1]; dst[2] = pk[2]; dst[3] = pk[3];

    // Single barrier per iteration (double buffer): writes of iteration it+2 to this
    // buffer are ordered behind barrier it+1, which every thread passes only after
    // its WMMA reads of iteration it completed -> safe, and lets WMMA (TRANS-class)
    // co-execute with the next chunk's VALU work.
    __syncthreads();

    // ---- stage 4 GEMM slice: 7 waves x (16 K-steps) of v_wmma_f32_16x16x32_f16 ----
    if (wave < 7) {
      const _Float16* bp = Bbase + (size_t)it * KCHUNK;
      __builtin_prefetch(bp + KCHUNK, 0, 1);   // stream next B chunk (L2-resident)
      const _Float16* ap0 = &Atile[buf][arow][aoff];
#pragma unroll
      for (int ks = 0; ks < KCHUNK / 32; ++ks) {
        v16h bfrag = *(const v16h*)(bp + ks * 32);          // contiguous: K pairs in VGPR order
        v16h afrag;
        const _Float16* ap = ap0 + ks * 32;
        ((uint4*)&afrag)[0] = *(const uint4*)(ap);          // K {0..7} / {16..23}
        ((uint4*)&afrag)[1] = *(const uint4*)(ap + 8);      // K {8..15} / {24..31}
        acc = __builtin_amdgcn_wmma_f32_16x16x32_f16(false, afrag, false, bfrag,
                                                     (short)0, acc, false, false);
      }
    }
  }

  // ---- epilogue: +b1a, ReLU, *W1b applied on C layout; reduce over N via LDS ----
  if (wave < 7) {
#pragma unroll
    for (int i = 0; i < 8; ++i) {
      int n  = lane & 15;
      int r  = wave * 16 + n;
      int mb = i + ((lane >> 4) << 3);
      float sc = 0.f;
      if (r < 100) sc = fmaxf(acc[i] + b1a[r], 0.f) * W1b[r];
      ep[wave][mb][n] = sc;
    }
  }
  __syncthreads();
  if (tid < 16) {
    float s = b1b[0];
    for (int w = 0; w < 7; ++w)
#pragma unroll
      for (int n = 0; n < 16; ++n) s += ep[w][tid][n];
    out[batch0 + tid] = 1.f / (1.f + __expf(-s));
  }
}

extern "C" void kernel_launch(void* const* d_in, const int* in_sizes, int n_in,
                              void* d_out, int out_size, void* d_ws, size_t ws_size,
                              hipStream_t stream) {
  const float* x   = (const float*)d_in[0];
  const float* w00 = (const float*)d_in[1];
  const float* b00 = (const float*)d_in[2];
  const float* W01 = (const float*)d_in[3];
  const float* b01 = (const float*)d_in[4];
  const float* W02 = (const float*)d_in[5];
  const float* b02 = (const float*)d_in[6];
  const float* W1a = (const float*)d_in[7];
  const float* b1a = (const float*)d_in[8];
  const float* W1b = (const float*)d_in[9];
  const float* b1b = (const float*)d_in[10];
  float* out = (float*)d_out;

  char* ws = (char*)d_ws;
  _Float16* W1ah = (_Float16*)ws;                                // 112*16384 halves = 3.67 MB
  float*    Mg   = (float*)(ws + (size_t)NPAD * KPAD * 2);       // 169 floats
  float*    bbg  = Mg + 256;                                     // 13 floats

  prep_small<<<1, 256, 0, stream>>>(W01, b01, W02, b02, Mg, bbg);
  prep_w1a<<<(NPAD * KPAD + 255) / 256, 256, 0, stream>>>(W1a, W1ah);
  fused_main<<<BSZ / 16, 256, 0, stream>>>(x, w00, b00, W1ah, Mg, bbg, b1a, W1b, b1b, out);
}
